// DecoderStructureWithAttention_77111842833036
// MI455X (gfx1250) — compile-verified, matrix operand order checked by reading, not run on hardware
//
#include <hip/hip_runtime.h>
#include <hip/hip_bf16.h>

typedef __attribute__((ext_vector_type(16))) _Float16 v16h;
typedef __attribute__((ext_vector_type(8)))  _Float16 v8h;
typedef __attribute__((ext_vector_type(8)))  float    v8f;
typedef __attribute__((ext_vector_type(4))) unsigned int u32x4;
typedef __attribute__((ext_vector_type(8))) int          i32x8;
typedef __attribute__((ext_vector_type(4))) int          i32x4;

union HU { v16h v; v8h h[2]; };

#define BB 64      // batch
#define PP 196     // pixels (14*14)
#define EE 512     // encoder dim
#define DD 512     // decoder dim
#define TT 256     // time steps (L-1)
#define LL 257
#define VV 1000

// ---------------------------------------------------------------------------
// WMMA GEMM with TDM-staged A tile:
//   C[M x N](f32) = A[M x K](f16 row-major) x W[N x K](f16 row-major)^T + bias
// One wave = one 16x16 tile. All 8 waves of a block share the same tm
// (requires tilesN % 8 == 0 and grid == tiles/8), so the block's 16 x K A tile
// is DMA'd into LDS once via tensor_load_to_lds (TDM), with LDS padding
// (4 DWORDs per 256 DWORDs) so row stride mod 64 banks is 4/12 -> conflict-free
// 16B-aligned ds_load_b128 reads. B streams from global (L2-resident weights).
// fcmode: masked prediction epilogue written straight to d_out (cols < 1000).
// ---------------------------------------------------------------------------
template <int KT>
__global__ void gemm_wmma_f16(const _Float16* __restrict__ A,
                              const _Float16* __restrict__ W,
                              const float* __restrict__ bias,
                              float* __restrict__ C, int N,
                              int fcmode, const int* __restrict__ dec_len,
                              int t, float* __restrict__ outp) {
  constexpr int ROWDW = (KT >> 1) + ((KT >> 9) << 2);  // padded LDS row (DWORDs)
  __shared__ alignas(16) _Float16 sA[16 * ROWDW * 2];

  int lane = (int)(threadIdx.x & 31);
  int waveInBlk = (int)(threadIdx.x >> 5);
  int tilesN = N >> 4;
  int tileBase = (int)blockIdx.x * 8;      // first tile of this block
  int tm = tileBase / tilesN;              // block-uniform (SGPR) M-tile
  int tn = (tileBase % tilesN) + waveInBlk;

  // ---- TDM: one async tensor DMA stages the 16 x KT f16 A tile into LDS ----
  if (threadIdx.x < 32) {
    unsigned long long ga =
        (unsigned long long)(const void*)(A + (size_t)tm * 16 * (size_t)KT);
    unsigned lds = (unsigned)(unsigned long long)(const void*)&sA[0];
    u32x4 g0;
    g0[0] = 1u;                                            // count=1, user D#
    g0[1] = lds;                                           // lds_addr
    g0[2] = (unsigned)ga;                                  // global_addr[31:0]
    g0[3] = (unsigned)((ga >> 32) & 0x01FFFFFFu) | (2u << 30);  // addr[56:32] | type=2
    i32x8 g1;
    g1[0] = (int)0x07D10000u;  // data_size=2B | pad_enable | pad_interval=256DW | pad_amount=4DW
    g1[1] = (int)(((unsigned)KT & 0xFFFFu) << 16);         // tensor_dim0[15:0]
    g1[2] = (int)((((unsigned)KT) >> 16) | (16u << 16));   // tensor_dim0[31:16] | tensor_dim1=16
    g1[3] = (int)(((unsigned)KT) << 16);                   // tile_dim0 = KT
    g1[4] = 16;                                            // tile_dim1 = 16
    g1[5] = KT;                                            // tensor_dim0_stride = KT
    g1[6] = 0;
    g1[7] = 0;
    i32x4 z4 = {0, 0, 0, 0};
#if defined(__clang_major__) && (__clang_major__ >= 23)
    i32x8 z8 = {0, 0, 0, 0, 0, 0, 0, 0};
    __builtin_amdgcn_tensor_load_to_lds(g0, g1, z4, z4, z8, 0);
#else
    __builtin_amdgcn_tensor_load_to_lds(g0, g1, z4, z4, 0);
#endif
    __builtin_amdgcn_s_wait_tensorcnt(0);
  }
  __syncthreads();

  int r16 = lane & 15;
  int kb  = (lane >> 4) << 3;                // 0 or 8 (ISA 16-bit 16x32 layout)
  const _Float16* wp = W + (size_t)(tn * 16 + r16) * (size_t)KT;
  v8f acc = {0.f, 0.f, 0.f, 0.f, 0.f, 0.f, 0.f, 0.f};
  for (int k0 = 0; k0 < KT; k0 += 32) {
    HU a, b;
    int dwa = (k0 + kb) >> 1;                // DWORD index in row (multiple of 4)
    int dwb = dwa + 8;
    int la = r16 * ROWDW + dwa + ((dwa >> 8) << 2);  // account for TDM padding
    int lb = r16 * ROWDW + dwb + ((dwb >> 8) << 2);
    a.h[0] = *(const v8h*)&sA[la * 2];
    a.h[1] = *(const v8h*)&sA[lb * 2];
    b.h[0] = *(const v8h*)(wp + k0 + kb);
    b.h[1] = *(const v8h*)(wp + k0 + 16 + kb);
    acc = __builtin_amdgcn_wmma_f32_16x16x32_f16(
        false, a.v, false, b.v, (short)0, acc, false, false);
  }

  int col = tn * 16 + r16;
  float bv = bias ? bias[col] : 0.0f;
  int rbase = tm * 16 + ((lane >> 4) << 3);  // C/D layout: lanes>=16 hold M+8
  if (fcmode) {
    // masked vocab-logit epilogue straight to d_out (rows are batch indices)
#pragma unroll
    for (int r = 0; r < 8; ++r) {
      int row = rbase + r;
      float m = (t < dec_len[row]) ? 1.0f : 0.0f;
      if (col < VV)
        outp[((size_t)row * TT + t) * VV + col] = m * (acc[r] + bv);
    }
  } else {
#pragma unroll
    for (int r = 0; r < 8; ++r)
      C[(size_t)(rbase + r) * (size_t)N + col] = acc[r] + bv;
  }
}

// ---------------------------------------------------------------------------
// Stable sort by caption length (descending), emit sort_ind/dec_len + outputs
// ---------------------------------------------------------------------------
__global__ void k_sort(const int* __restrict__ cap_lens,  // [64,1]
                       const int* __restrict__ caps,      // [64,257]
                       int* __restrict__ sort_ind, int* __restrict__ dec_len,
                       float* __restrict__ out_caps, float* __restrict__ out_sortind) {
  __shared__ int s_ind[BB];
  int tid = threadIdx.x;
  if (tid == 0) {
    bool used[BB];
    for (int i = 0; i < BB; ++i) used[i] = false;
    for (int i = 0; i < BB; ++i) {
      int best = 0, bestlen = -2147483647;
      for (int j = 0; j < BB; ++j)
        if (!used[j] && cap_lens[j] > bestlen) { bestlen = cap_lens[j]; best = j; }
      used[best] = true;
      s_ind[i] = best;
    }
  }
  __syncthreads();
  int b = tid;   // 64 threads
  int src = s_ind[b];
  sort_ind[b] = src;
  dec_len[b]  = cap_lens[src] - 1;
  out_sortind[b] = (float)src;
  for (int l = 0; l < LL; ++l)
    out_caps[b * LL + l] = (float)caps[src * LL + l];
}

// ---------------------------------------------------------------------------
// Weight conversion / concatenation / padding to f16
// ---------------------------------------------------------------------------
__global__ void k_prep(const float* __restrict__ dec_att_w, const float* __restrict__ dec_att_b,
                       const float* __restrict__ f_beta_w,  const float* __restrict__ f_beta_b,
                       const float* __restrict__ enc_att_w,
                       const float* __restrict__ init_h_w,  const float* __restrict__ init_h_b,
                       const float* __restrict__ init_c_w,  const float* __restrict__ init_c_b,
                       const float* __restrict__ W_ih, const float* __restrict__ b_ih,
                       const float* __restrict__ W_hh, const float* __restrict__ b_hh,
                       const float* __restrict__ fc_w, const float* __restrict__ fc_b,
                       _Float16* __restrict__ dg_w, float* __restrict__ dg_b,
                       _Float16* __restrict__ enc_att_w_h,
                       _Float16* __restrict__ hc_w, float* __restrict__ hc_b,
                       _Float16* __restrict__ W_cat, float* __restrict__ b_cat,
                       _Float16* __restrict__ fc_pad, float* __restrict__ fc_b_pad) {
  const int TOT = 2048 * 1536;
  int stride = (int)(gridDim.x * blockDim.x);
  for (int i = (int)(blockIdx.x * blockDim.x + threadIdx.x); i < TOT; i += stride) {
    { // W_cat[2048][1536] = [W_ih | W_hh]
      int n = i / 1536, k = i - n * 1536;
      float v = (k < 1024) ? W_ih[n * 1024 + k] : W_hh[n * 512 + (k - 1024)];
      W_cat[i] = (_Float16)v;
    }
    if (i < 1024 * 512) {
      int r = i >> 9;
      dg_w[i]   = (_Float16)((r < 512)  ? dec_att_w[i] : f_beta_w[i - 512 * 512]);
      hc_w[i]   = (_Float16)((r < 512)  ? init_h_w[i]  : init_c_w[i - 512 * 512]);
      fc_pad[i] = (_Float16)((r < 1000) ? fc_w[i] : 0.0f);
    }
    if (i < 512 * 512) enc_att_w_h[i] = (_Float16)enc_att_w[i];
    if (i < 2048) b_cat[i] = b_ih[i] + b_hh[i];
    if (i < 1024) {
      dg_b[i]     = (i < 512)  ? dec_att_b[i] : f_beta_b[i - 512];
      hc_b[i]     = (i < 512)  ? init_h_b[i]  : init_c_b[i - 512];
      fc_b_pad[i] = (i < 1000) ? fc_b[i] : 0.0f;
    }
  }
}

// Sorted encoder features -> f16 (A operand of att1 GEMM)
__global__ void k_gather_enc(const float* __restrict__ encoder_out,
                             const int* __restrict__ sort_ind,
                             _Float16* __restrict__ enc_h) {
  const int TOT = BB * PP * EE;
  int stride = (int)(gridDim.x * blockDim.x);
  for (int i = (int)(blockIdx.x * blockDim.x + threadIdx.x); i < TOT; i += stride) {
    int b = i / (PP * EE);
    int rest = i - b * (PP * EE);
    enc_h[i] = (_Float16)encoder_out[(size_t)sort_ind[b] * (PP * EE) + rest];
  }
}

// Mean pool over pixels -> f16 (A operand of h0/c0 GEMM)
__global__ void k_mean(const float* __restrict__ encoder_out,
                       const int* __restrict__ sort_ind,
                       _Float16* __restrict__ mean_h) {
  int b = blockIdx.x;
  const float* base = encoder_out + (size_t)sort_ind[b] * (PP * EE);
  for (int e = threadIdx.x; e < EE; e += blockDim.x) {
    float s = 0.f;
    for (int p = 0; p < PP; ++p) s += base[p * EE + e];
    mean_h[b * EE + e] = (_Float16)(s * (1.0f / (float)PP));
  }
}

__global__ void k_init_hc(const float* __restrict__ hc0,
                          float* __restrict__ h, float* __restrict__ c,
                          _Float16* __restrict__ h_h) {
  int b = blockIdx.x;
  for (int d = threadIdx.x; d < DD; d += blockDim.x) {
    float hv = hc0[b * 1024 + d];
    float cv = hc0[b * 1024 + 512 + d];
    h[b * DD + d] = hv;
    c[b * DD + d] = cv;
    h_h[b * DD + d] = (_Float16)hv;
  }
}

// ---------------------------------------------------------------------------
// Attention step: e = relu(att1+att2)·w, softmax over P, awe, gate, x assembly
// ---------------------------------------------------------------------------
__global__ void k_attention(const float* __restrict__ att1,       // [B*P*A]
                            const float* __restrict__ att2gate,   // [B*1024]
                            const float* __restrict__ full_att_w, // [A]
                            const float* __restrict__ full_att_b, // [1]
                            const float* __restrict__ encoder_out,// original [B,P,E]
                            const int* __restrict__ sort_ind,
                            const int* __restrict__ dec_len,
                            const int* __restrict__ caps,         // original [B,L]
                            const float* __restrict__ emb,        // [V,M]
                            const _Float16* __restrict__ h_h,     // [B*D]
                            _Float16* __restrict__ x_h,           // [B*1536]
                            float* __restrict__ out_alphas,
                            int t) {
  __shared__ float s_e[PP];
  __shared__ float s_red[256];
  int b = blockIdx.x;
  int tid = threadIdx.x;
  int lane = tid & 31;
  int wave = tid >> 5;
  const float* a1 = att1 + (size_t)b * PP * DD;
  const float* a2 = att2gate + b * 1024;

  for (int p = wave; p < PP; p += 8) {
    float partial = 0.f;
    for (int a = lane; a < DD; a += 32) {
      float v = a1[p * DD + a] + a2[a];
      v = v > 0.f ? v : 0.f;
      partial += v * full_att_w[a];
    }
    for (int off = 16; off > 0; off >>= 1)
      partial += __shfl_xor(partial, off, 32);
    if (lane == 0) s_e[p] = partial + full_att_b[0];
  }
  __syncthreads();

  float v = (tid < PP) ? s_e[tid] : -3.0e38f;
  s_red[tid] = v;
  __syncthreads();
  for (int s = 128; s > 0; s >>= 1) {
    if (tid < s) s_red[tid] = fmaxf(s_red[tid], s_red[tid + s]);
    __syncthreads();
  }
  float mx = s_red[0];
  __syncthreads();
  float ev = (tid < PP) ? __expf(s_e[tid] - mx) : 0.f;
  s_red[tid] = ev;
  __syncthreads();
  for (int s = 128; s > 0; s >>= 1) {
    if (tid < s) s_red[tid] += s_red[tid + s];
    __syncthreads();
  }
  float inv = 1.0f / s_red[0];
  __syncthreads();
  if (tid < PP) s_e[tid] = ev * inv;   // alpha
  __syncthreads();

  int src = sort_ind[b];
  float m = (t < dec_len[b]) ? 1.0f : 0.0f;
  for (int p = tid; p < PP; p += 256)
    out_alphas[((size_t)b * TT + t) * PP + p] = s_e[p] * m;

  const float* encb = encoder_out + (size_t)src * PP * EE;
  int cap = caps[src * LL + t];
  const float* embrow = emb + (size_t)cap * 512;
  _Float16* xb = x_h + (size_t)b * 1536;
  for (int e = tid; e < EE; e += 256) {
    float awe = 0.f;
    for (int p = 0; p < PP; ++p) awe += s_e[p] * encb[p * EE + e];
    float g = 1.0f / (1.0f + __expf(-a2[512 + e]));
    xb[e]        = (_Float16)embrow[e];     // emb_t
    xb[512 + e]  = (_Float16)(g * awe);     // gate * awe
    xb[1024 + e] = h_h[b * DD + e];         // h (for W_hh block of fused GEMM)
  }
}

// ---------------------------------------------------------------------------
// LSTM pointwise: gates -> (h_new, c_new), masked carry, f16 mirrors
// ---------------------------------------------------------------------------
__global__ void k_lstm(const float* __restrict__ gates,   // [B*2048]
                       const float* __restrict__ h_in, const float* __restrict__ c_in,
                       float* __restrict__ h_out, float* __restrict__ c_out,
                       _Float16* __restrict__ h_h, _Float16* __restrict__ h_new_h,
                       const int* __restrict__ dec_len, int t) {
  int b = blockIdx.x;
  float m = (t < dec_len[b]) ? 1.0f : 0.0f;
  const float* g = gates + (size_t)b * 2048;
  for (int d = threadIdx.x; d < DD; d += blockDim.x) {
    float ig = 1.f / (1.f + __expf(-g[d]));
    float fg = 1.f / (1.f + __expf(-g[512 + d]));
    float gg = tanhf(g[1024 + d]);
    float og = 1.f / (1.f + __expf(-g[1536 + d]));
    float cn = fg * c_in[b * DD + d] + ig * gg;
    float hn = og * tanhf(cn);
    float h2 = m * hn + (1.f - m) * h_in[b * DD + d];
    float c2 = m * cn + (1.f - m) * c_in[b * DD + d];
    h_out[b * DD + d] = h2;
    c_out[b * DD + d] = c2;
    h_h[b * DD + d]     = (_Float16)h2;   // next-step WMMA A operand
    h_new_h[b * DD + d] = (_Float16)hn;   // fc GEMM A operand
  }
}

// ---------------------------------------------------------------------------
extern "C" void kernel_launch(void* const* d_in, const int* in_sizes, int n_in,
                              void* d_out, int out_size, void* d_ws, size_t ws_size,
                              hipStream_t stream) {
  (void)in_sizes; (void)n_in; (void)out_size; (void)ws_size;
  const float* encoder_out = (const float*)d_in[0];
  const int*   caps        = (const int*)d_in[1];
  const int*   cap_lens    = (const int*)d_in[2];
  const float* emb         = (const float*)d_in[3];
  const float* enc_att_w   = (const float*)d_in[4];
  const float* enc_att_b   = (const float*)d_in[5];
  const float* dec_att_w   = (const float*)d_in[6];
  const float* dec_att_b   = (const float*)d_in[7];
  const float* full_att_w  = (const float*)d_in[8];
  const float* full_att_b  = (const float*)d_in[9];
  const float* W_ih        = (const float*)d_in[10];
  const float* b_ih        = (const float*)d_in[11];
  const float* W_hh        = (const float*)d_in[12];
  const float* b_hh        = (const float*)d_in[13];
  const float* init_h_w    = (const float*)d_in[14];
  const float* init_h_b    = (const float*)d_in[15];
  const float* init_c_w    = (const float*)d_in[16];
  const float* init_c_b    = (const float*)d_in[17];
  const float* f_beta_w    = (const float*)d_in[18];
  const float* f_beta_b    = (const float*)d_in[19];
  const float* fc_w        = (const float*)d_in[20];
  const float* fc_b        = (const float*)d_in[21];

  float* out = (float*)d_out;
  const size_t OFF_CAPS   = (size_t)BB * TT * VV;           // 16,384,000
  const size_t OFF_ALPHAS = OFF_CAPS + (size_t)BB * LL;     // 16,400,448
  const size_t OFF_SORT   = OFF_ALPHAS + (size_t)BB * TT * PP;

  char* ws = (char*)d_ws;
  size_t off = 0;
  auto alloc = [&](size_t bytes) -> void* {
    off = (off + 255) & ~(size_t)255;
    void* p = ws + off;
    off += bytes;
    return p;
  };

  int*      sort_ind = (int*)alloc(BB * 4);
  int*      dec_len  = (int*)alloc(BB * 4);
  _Float16* dg_w     = (_Float16*)alloc((size_t)1024 * 512 * 2);
  float*    dg_b     = (float*)alloc(1024 * 4);
  _Float16* enc_w_h  = (_Float16*)alloc((size_t)512 * 512 * 2);
  _Float16* hc_w     = (_Float16*)alloc((size_t)1024 * 512 * 2);
  float*    hc_b     = (float*)alloc(1024 * 4);
  _Float16* W_cat    = (_Float16*)alloc((size_t)2048 * 1536 * 2);
  float*    b_cat    = (float*)alloc(2048 * 4);
  _Float16* fc_pad   = (_Float16*)alloc((size_t)1024 * 512 * 2);
  float*    fc_b_pad = (float*)alloc(1024 * 4);
  _Float16* enc_h    = (_Float16*)alloc((size_t)BB * PP * EE * 2);
  float*    att1     = (float*)alloc((size_t)BB * PP * DD * 4);
  _Float16* mean_h   = (_Float16*)alloc((size_t)BB * EE * 2);
  float*    hc0      = (float*)alloc((size_t)BB * 1024 * 4);
  float*    h_buf0   = (float*)alloc((size_t)BB * DD * 4);
  float*    h_buf1   = (float*)alloc((size_t)BB * DD * 4);
  float*    c_buf0   = (float*)alloc((size_t)BB * DD * 4);
  float*    c_buf1   = (float*)alloc((size_t)BB * DD * 4);
  _Float16* h_h      = (_Float16*)alloc((size_t)BB * DD * 2);
  _Float16* h_new_h  = (_Float16*)alloc((size_t)BB * DD * 2);
  float*    att2gate = (float*)alloc((size_t)BB * 1024 * 4);
  _Float16* x_h      = (_Float16*)alloc((size_t)BB * 1536 * 2);
  float*    gatesbuf = (float*)alloc((size_t)BB * 2048 * 4);

  // GEMM launcher: tiles/8 blocks of 8 waves; tilesN % 8 == 0 for all shapes
  auto gemm512 = [&](const _Float16* A, const _Float16* W, const float* bias,
                     float* C, int M, int N) {
    int blocks = ((M >> 4) * (N >> 4)) / 8;
    gemm_wmma_f16<512><<<blocks, 256, 0, stream>>>(A, W, bias, C, N,
                                                   0, dec_len, 0, (float*)nullptr);
  };

  // ---- setup phase ----
  k_sort<<<1, 64, 0, stream>>>(cap_lens, caps, sort_ind, dec_len,
                               out + OFF_CAPS, out + OFF_SORT);
  k_prep<<<2048, 256, 0, stream>>>(dec_att_w, dec_att_b, f_beta_w, f_beta_b,
                                   enc_att_w, init_h_w, init_h_b, init_c_w, init_c_b,
                                   W_ih, b_ih, W_hh, b_hh, fc_w, fc_b,
                                   dg_w, dg_b, enc_w_h, hc_w, hc_b,
                                   W_cat, b_cat, fc_pad, fc_b_pad);
  k_gather_enc<<<8192, 256, 0, stream>>>(encoder_out, sort_ind, enc_h);
  k_mean<<<BB, 256, 0, stream>>>(encoder_out, sort_ind, mean_h);
  gemm512(mean_h, hc_w, hc_b, hc0, 64, 1024);                  // h0 | c0
  k_init_hc<<<BB, 256, 0, stream>>>(hc0, h_buf0, c_buf0, h_h);
  gemm512(enc_h, enc_w_h, enc_att_b, att1, BB * PP, 512);      // att1 (big GEMM)

  // ---- 256 recurrent steps ----
  float* hp[2] = {h_buf0, h_buf1};
  float* cp[2] = {c_buf0, c_buf1};
  int ping = 0;
  for (int t = 0; t < TT; ++t) {
    gemm512(h_h, dg_w, dg_b, att2gate, 64, 1024);              // att2 | gate-pre
    k_attention<<<BB, 256, 0, stream>>>(att1, att2gate, full_att_w, full_att_b,
                                        encoder_out, sort_ind, dec_len, caps, emb,
                                        h_h, x_h, out + OFF_ALPHAS, t);
    // LSTM gates: M=64, N=2048, K=1536 (fused [W_ih | W_hh])
    gemm_wmma_f16<1536><<<64, 256, 0, stream>>>(x_h, W_cat, b_cat, gatesbuf, 2048,
                                                0, dec_len, 0, (float*)nullptr);
    k_lstm<<<BB, 256, 0, stream>>>(gatesbuf, hp[ping], cp[ping],
                                   hp[1 - ping], cp[1 - ping],
                                   h_h, h_new_h, dec_len, t);
    // vocab logits with fused masked write-out straight to d_out
    gemm_wmma_f16<512><<<32, 256, 0, stream>>>(h_new_h, fc_pad, fc_b_pad, hc0, 1024,
                                               1, dec_len, t, out);
    ping ^= 1;
  }
}
